// LSTMModel_78245714198762
// MI455X (gfx1250) — matrix operand order repaired and checked by reading, not compiled, not run on hardware
//
#include <hip/hip_runtime.h>
#include <hip/hip_bf16.h>

typedef __attribute__((ext_vector_type(16))) _Float16 v16h;
typedef __attribute__((ext_vector_type(8)))  float    v8f;

#define B_    64
#define T_    64
#define KC_   32
#define H_    512
#define NC_   10000
#define G4_   2048          // 4*H
#define DEMO_ 16
#define KP_   1056          // 2H+DEMO (=1040) padded to multiple of 32

// ---------------------------------------------------------------------------
// WMMA fragment helpers (contiguous k = 16*(lane/16) + s in A and B, wave32)
// ---------------------------------------------------------------------------

// A-matrix 16x32 f16: lanes 0-15 hold M=lane, K in {0..15};
// lanes 16-31 hold M=lane-16, K in {16..31}; pairs are K-consecutive,
// so each half assembles from two 16B loads (compiler-proven b128s).
__device__ __forceinline__ int a_koff(int lane, int q) {
  int base = (lane & 16) ? 16 : 0;
  return base + 2 * q;                         // q=0..7 pair start offsets
}

// B-matrix 32x16 f16: lane&15 = N; lanes 0-15 hold K=0..15, lanes 16-31
// K=16..31 => 16 contiguous halves per lane: one aligned 32B load.
__device__ __forceinline__ v16h bfrag_load(const _Float16* row_kb, int lane) {
  return *(const v16h*)(row_kb + ((lane & 16) ? 16 : 0));
}

__device__ __forceinline__ v8f wmma_f16(v16h a, v16h b, v8f c) {
  // (neg_a, A, neg_b, B, c_mod, C, reuse_a, reuse_b)
  return __builtin_amdgcn_wmma_f32_16x16x32_f16(false, a, false, b, (short)0, c,
                                                false, false);
}

// Branch-free transcendentals on the v_exp_f32 / v_rcp_f32 TRANS pipes.
// Saturation is exact: exp2->0/inf gives sigm->{0,1}, tanh->{-1,1}.
#define LOG2E_ 1.4426950408889634f
__device__ __forceinline__ float sigm(float x) {
  float e = __builtin_amdgcn_exp2f(-LOG2E_ * x);          // exp(-x)
  return __builtin_amdgcn_rcpf(1.0f + e);
}
__device__ __forceinline__ float ftanh(float x) {
  float e = __builtin_amdgcn_exp2f((2.0f * LOG2E_) * x);  // exp(2x)
  return 1.0f - 2.0f * __builtin_amdgcn_rcpf(e + 1.0f);
}

// ---------------------------------------------------------------------------
// Generic WMMA GEMM:  C[M,N] = act( Agather[M,K] * Bw[N,K]^T + bias )
// MODE 0: A row r at A + r*lda (flat)
// MODE 1: r -> (b=r%64, t=r/64), A row = A + (b*T + t)*lda          (seq fwd)
// MODE 2: same but t -> len[b]-1-t, zero row when t >= len[b]       (seq rev)
// NT n-tiles per wave (A-fragment reuse, cuts L2 A-panel re-reads by NT).
// CT: store C transposed as [t][N][64] (one v8f store per tile per lane).
// ---------------------------------------------------------------------------
template <int MODE, int NT, bool AF16, bool RELU, bool OUTF16, bool CT>
__global__ void gemm_wmma_k(const void* __restrict__ Aptr, int lda,
                            const _Float16* __restrict__ Bw,   // [N,K] row-major
                            const float* __restrict__ bias,
                            void* __restrict__ Cptr, int M, int N, int K,
                            const int* __restrict__ lengths) {
  const int lane   = threadIdx.x & 31;
  const int wid    = threadIdx.x >> 5;          // 4 waves / block
  const int mtile  = blockIdx.y * 4 + wid;
  const int ntile0 = blockIdx.x * NT;
  const int mloc   = lane & 15;
  if (mtile * 16 >= M) return;
  const int gm = mtile * 16 + mloc;             // A row this lane feeds

  const float*    Af  = (const float*)Aptr;
  const _Float16* Ah  = (const _Float16*)Aptr;
  const float*    arf = nullptr;
  const _Float16* arh = nullptr;
  bool valid = true;
  if (MODE == 0) {
    if (AF16) arh = Ah + (size_t)gm * lda; else arf = Af + (size_t)gm * lda;
  } else {
    int b = gm & (B_ - 1);
    int t = gm >> 6;
    int st = t;
    if (MODE == 2) {
      int L = lengths[b];
      valid = (t < L);
      st    = valid ? (L - 1 - t) : 0;
    }
    size_t ro = ((size_t)b * T_ + st) * (size_t)lda;
    if (AF16) arh = Ah + ro; else arf = Af + ro;
  }

  v8f acc[NT];
#pragma unroll
  for (int u = 0; u < NT; ++u) acc[u] = (v8f){0.f,0.f,0.f,0.f,0.f,0.f,0.f,0.f};

  for (int kb = 0; kb < K; kb += 32) {
    v16h a;
    if (valid) {
#pragma unroll
      for (int q = 0; q < 8; ++q) {
        int k = a_koff(lane, q);
        if (AF16) { a[2*q] = arh[kb + k];           a[2*q+1] = arh[kb + k + 1]; }
        else      { a[2*q] = (_Float16)arf[kb + k]; a[2*q+1] = (_Float16)arf[kb + k + 1]; }
      }
    } else {
#pragma unroll
      for (int q = 0; q < 16; ++q) a[q] = (_Float16)0.f;
    }
#pragma unroll
    for (int u = 0; u < NT; ++u) {
      const _Float16* brow = Bw + (size_t)((ntile0 + u) * 16 + mloc) * K + kb;
      v16h bf = bfrag_load(brow, lane);
      if (u == 0) __builtin_prefetch((const void*)(brow + 64), 0, 1);
      acc[u] = wmma_f16(a, bf, acc[u]);
    }
  }

#pragma unroll
  for (int u = 0; u < NT; ++u) {
    const int n  = (ntile0 + u) * 16 + mloc;         // C: N = lane&15
    const float bv = bias ? bias[n] : 0.f;
    if (CT) {
      // C[t][n][b]: each lane's 8 M-values are 8 consecutive b's -> one v8f
      int tq = (mtile * 16) >> 6;
      int b0 = ((mtile * 16) & 63) + ((lane < 16) ? 0 : 8);
      v8f ov;
#pragma unroll
      for (int r = 0; r < 8; ++r) ov[r] = acc[u][r] + bv;
      *(v8f*)((float*)Cptr + ((size_t)tq * N + n) * 64 + b0) = ov;
    } else {
#pragma unroll
      for (int r = 0; r < 8; ++r) {                  // C: M = r (+8 hi lanes)
        int m = mtile * 16 + ((lane < 16) ? r : r + 8);
        if (m >= M) continue;
        float v = acc[u][r] + bv;
        if (RELU) v = fmaxf(v, 0.f);
        if (OUTF16) ((_Float16*)Cptr)[(size_t)m * N + n] = (_Float16)v;
        else        ((float*)Cptr)[(size_t)m * N + n]    = v;
      }
    }
  }
}

// ---------------------------------------------------------------------------
// Persistent LSTM recurrence, batch-row x column-slice parallel.
// grid = 32 blocks: (dir:2) x (row-tile:4) x (col-slice:4); block = 256 = 8 waves.
// Each wave owns one 16-wide h-column tile: its 4 accumulators are exactly the
// i/f/g/o gate tiles for those columns -> lane-local nonlinearity update.
// h state: f16, double-buffered in L2-resident global (per group [2][16][512]);
// the read buffer is staged into LDS once per step so all 8 waves' A-fragments
// come from ds_load_b128 instead of 8x-redundant global loads.
// c state: f32 in LDS (column owner only). One monotonic-counter split barrier
// per step across the 4 col-slice blocks of a group (double buffering makes a
// single barrier sufficient). xg layout: [t][4H][B] with bih+bhh pre-folded.
// ---------------------------------------------------------------------------
__global__ void __launch_bounds__(256)
lstm_col_k(const float* __restrict__ xg_f, const float* __restrict__ xg_b,
           const _Float16* __restrict__ whh_f, const _Float16* __restrict__ whh_b,
           _Float16* __restrict__ hbuf,     // [8 groups][2][16][H_]
           int* __restrict__ cnt,           // [8] monotonic arrival counters
           float* __restrict__ out0,        // [b][t][2H] or null
           float* __restrict__ hTf, float* __restrict__ hTb,
           const int* __restrict__ lengths, int writeOut) {
  const int tid   = threadIdx.x;
  const int lane  = tid & 31;
  const int w     = tid >> 5;               // wave 0..7
  const int nl    = lane & 15;
  const int slice = blockIdx.x & 3;
  const int mtile = (blockIdx.x >> 2) & 3;
  const int dir   = blockIdx.x >> 4;
  const int g     = dir * 4 + mtile;
  const int m0    = mtile * 16;
  const int ct    = slice * 8 + w;          // col-tile 0..31
  const int j     = ct * 16 + nl;           // h column this lane owns

  const float*    xg  = dir ? xg_b  : xg_f;
  const _Float16* whh = dir ? whh_b : whh_f;
  float*          hT  = dir ? hTb   : hTf;
  _Float16*       hb  = hbuf + (size_t)g * (2 * 16 * H_);

  __shared__ _Float16 hS[16][H_];    // staged read buffer (16 KB)
  __shared__ float    cC[16][128];   // owned cell state (8 KB)
  __shared__ int      lenS[16];
  for (int idx = tid; idx < 16 * 128; idx += 256) cC[idx >> 7][idx & 127] = 0.f;
  if (tid < 16) lenS[tid] = lengths[m0 + tid];
  __syncthreads();

  for (int t = 0; t < T_; ++t) {
    const _Float16* R = hb + (size_t)(t & 1) * (16 * H_);        // read buf
    _Float16*       W = hb + (size_t)((t + 1) & 1) * (16 * H_);  // write buf

    // stage h(t-1) panel into LDS (coalesced 16B chunks)
    {
      const uint4* src = (const uint4*)R;
      uint4*       dst = (uint4*)&hS[0][0];
      for (int idx = tid; idx < (16 * H_ * 2) / 16; idx += 256) dst[idx] = src[idx];
    }
    __syncthreads();

    // ---- gates = xg[t] + h @ Whh^T ; acc[G] = gate G (i,f,g,o) ----
    v8f acc[4];
    {
      const float* xgt = xg + (size_t)t * (G4_ * B_);
#pragma unroll
      for (int G = 0; G < 4; ++G) {
        int row = G * H_ + j;                // gate column in [0,2048)
        acc[G] = *(const v8f*)(xgt + (size_t)row * B_ + m0 + ((lane < 16) ? 0 : 8));
      }
    }
    for (int kb = 0; kb < H_; kb += 32) {
      v16h a;
#pragma unroll
      for (int q = 0; q < 8; ++q) {
        int k    = a_koff(lane, q);
        a[2*q]   = hS[nl][kb + k];
        a[2*q+1] = hS[nl][kb + k + 1];
      }
#pragma unroll
      for (int G = 0; G < 4; ++G) {
        int nrow = G * H_ + ct * 16 + nl;    // Whh row = gate column
        v16h bf  = bfrag_load(whh + (size_t)nrow * H_ + kb, lane);
        acc[G]   = wmma_f16(a, bf, acc[G]);
      }
    }

    // ---- lane-local LSTM cell update on owned columns ----
#pragma unroll
    for (int r = 0; r < 8; ++r) {
      int  li = (lane < 16) ? r : r + 8;
      int  bg = m0 + li;
      bool mv = (t < lenS[li]);
      float iv = sigm(acc[0][r]);
      float fv = sigm(acc[1][r]);
      float gv = ftanh(acc[2][r]);
      float ov = sigm(acc[3][r]);
      float co = cC[li][j & 127];
      float cn = fv * co + iv * gv;
      float hn = ov * ftanh(cn);
      float ho = (float)hS[li][j];
      float cu = mv ? cn : co;
      float hu = mv ? hn : ho;
      cC[li][j & 127]          = cu;
      W[(size_t)li * H_ + j]   = (_Float16)hu;
      if (writeOut) {
        if (dir == 0)
          out0[((size_t)bg * T_ + t) * (2 * H_) + j] = hu;
        else if (mv)   // written pre-reversed; padded tail stays zero
          out0[((size_t)bg * T_ + (lenS[li] - 1 - t)) * (2 * H_) + H_ + j] = hu;
      }
    }

    // ---- split barrier across the 4 col-slice blocks of this group ----
    __threadfence();
    __syncthreads();
    if (tid == 0) {
      atomicAdd(&cnt[g], 1);
      const int target = (t + 1) * 4;
      while (__hip_atomic_load(&cnt[g], __ATOMIC_RELAXED,
                               __HIP_MEMORY_SCOPE_AGENT) < target)
        __builtin_amdgcn_s_sleep(2);
    }
    __syncthreads();
    __threadfence();
  }

  // final hidden state: T_ is even -> ends in buffer 0; own columns only
  if (hT) {
    const _Float16* F = hb;
    for (int idx = tid; idx < 16 * 128; idx += 256) {
      int r = idx >> 7, jj = slice * 128 + (idx & 127);
      hT[(size_t)(m0 + r) * H_ + jj] = (float)F[(size_t)r * H_ + jj];
    }
  }
}

// ---------------------------------------------------------------------------
// code_W transpose [H][NC] -> [NC][H] so the embedding gather is coalesced
// ---------------------------------------------------------------------------
__global__ void trans_cw_k(const float* __restrict__ in, float* __restrict__ out) {
  __shared__ float tile[16][17];
  int c0 = blockIdx.x * 16, h0 = blockIdx.y * 16;
  int tx = threadIdx.x, ty = threadIdx.y;
  tile[ty][tx] = in[(size_t)(h0 + ty) * NC_ + c0 + tx];   // coalesced in c
  __syncthreads();
  out[(size_t)(c0 + ty) * H_ + h0 + tx] = tile[tx][ty];   // coalesced in h
}

// ---------------------------------------------------------------------------
// Embedding: multi-hot (set semantics) gather of code rows + bias, masked
// ---------------------------------------------------------------------------
__global__ void embed_k(const int* __restrict__ codes, const int* __restrict__ lengths,
                        const float* __restrict__ cWT,   // [NC][H]
                        const float* __restrict__ cb,
                        float* __restrict__ x0) {
  int bt = blockIdx.x;
  int b = bt >> 6, t = bt & 63;
  __shared__ int sc[KC_];
  __shared__ int keep[KC_];
  if (threadIdx.x < KC_) sc[threadIdx.x] = codes[(size_t)bt * KC_ + threadIdx.x];
  __syncthreads();
  if (threadIdx.x < KC_) {
    int c = sc[threadIdx.x], k = 1;
    for (int jx = 0; jx < (int)threadIdx.x; ++jx)
      if (sc[jx] == c) { k = 0; break; }
    keep[threadIdx.x] = k;
  }
  __syncthreads();
  bool valid = t < lengths[b];
  for (int h = threadIdx.x; h < H_; h += blockDim.x) {
    float acc = 0.f;
    if (valid) {
      acc = cb[h];
      for (int jx = 0; jx < KC_; ++jx)
        if (keep[jx]) acc += cWT[(size_t)sc[jx] * H_ + h];   // stride-1 rows
    }
    x0[(size_t)bt * H_ + h] = acc;   // [b][t][H], bt = b*64 + t
  }
}

// ---------------------------------------------------------------------------
// Small utility kernels
// ---------------------------------------------------------------------------
__global__ void cvt_f16_k(const float* __restrict__ in, _Float16* __restrict__ out, int n) {
  int i = blockIdx.x * blockDim.x + threadIdx.x;
  if (i < n) out[i] = (_Float16)in[i];
}

__global__ void bsum_k(const float* __restrict__ a, const float* __restrict__ b,
                       float* __restrict__ o, int n) {
  int i = blockIdx.x * blockDim.x + threadIdx.x;
  if (i < n) o[i] = a[i] + b[i];
}

__global__ void w1pad_k(const float* __restrict__ W1, _Float16* __restrict__ out) {
  int i = blockIdx.x * blockDim.x + threadIdx.x;
  if (i >= H_ * KP_) return;
  int r = i / KP_, k = i % KP_;
  out[i] = (k < 2 * H_ + DEMO_) ? (_Float16)W1[(size_t)r * (2 * H_ + DEMO_) + k]
                                : (_Float16)0.f;
}

__global__ void comb_k(const float* __restrict__ hTf, const float* __restrict__ hTb,
                       const float* __restrict__ demo, _Float16* __restrict__ out) {
  int i = blockIdx.x * blockDim.x + threadIdx.x;
  if (i >= B_ * KP_) return;
  int b = i / KP_, c = i % KP_;
  float v;
  if (c < H_)                  v = hTf[b * H_ + c];
  else if (c < 2 * H_)         v = hTb[b * H_ + (c - H_)];
  else if (c < 2 * H_ + DEMO_) v = demo[b * DEMO_ + (c - 2 * H_)];
  else                         v = 0.f;
  out[i] = (_Float16)v;
}

// ---------------------------------------------------------------------------
extern "C" void kernel_launch(void* const* d_in, const int* in_sizes, int n_in,
                              void* d_out, int out_size, void* d_ws, size_t ws_size,
                              hipStream_t stream) {
  (void)in_sizes; (void)n_in; (void)out_size; (void)ws_size;

  const float* demo   = (const float*)d_in[0];
  const int*   codes  = (const int*)d_in[1];
  const int*   lens   = (const int*)d_in[2];
  const float* code_W = (const float*)d_in[3];
  const float* code_b = (const float*)d_in[4];
  const float* W1     = (const float*)d_in[5];
  const float* b1     = (const float*)d_in[6];
  const float* W2     = (const float*)d_in[7];
  const float* b2     = (const float*)d_in[8];
  // order: 0=l0f, 1=l0b, 2=l1f, 3=l1b
  const float* WihA[4] = {(const float*)d_in[9],  (const float*)d_in[13],
                          (const float*)d_in[17], (const float*)d_in[21]};
  const float* WhhA[4] = {(const float*)d_in[10], (const float*)d_in[14],
                          (const float*)d_in[18], (const float*)d_in[22]};
  const float* bihA[4] = {(const float*)d_in[11], (const float*)d_in[15],
                          (const float*)d_in[19], (const float*)d_in[23]};
  const float* bhhA[4] = {(const float*)d_in[12], (const float*)d_in[16],
                          (const float*)d_in[20], (const float*)d_in[24]};
  const int wihK[4] = {H_, H_, 2 * H_, 2 * H_};

  // ---- carve workspace ----
  char*  ws  = (char*)d_ws;
  size_t off = 0;
  auto take = [&](size_t bytes) -> char* {
    char* p = ws + off;
    off = (off + bytes + 255) & ~(size_t)255;
    return p;
  };
  _Float16* wih_h[4];
  _Float16* whh_h[4];
  float*    bsum[4];
  for (int i = 0; i < 4; ++i) {
    wih_h[i] = (_Float16*)take((size_t)G4_ * wihK[i] * 2);
    whh_h[i] = (_Float16*)take((size_t)G4_ * H_ * 2);
    bsum[i]  = (float*)take((size_t)G4_ * 4);
  }
  _Float16* w1p   = (_Float16*)take((size_t)H_ * KP_ * 2);
  _Float16* w2h   = (_Float16*)take((size_t)NC_ * H_ * 2);
  float*    cWT   = (float*)take((size_t)NC_ * H_ * 4);        // [NC][H]
  float*    x0    = (float*)take((size_t)B_ * T_ * H_ * 4);
  float*    xgf   = (float*)take((size_t)T_ * G4_ * B_ * 4);   // [t][4H][B]
  float*    xgb   = (float*)take((size_t)T_ * G4_ * B_ * 4);
  float*    out0  = (float*)take((size_t)B_ * T_ * 2 * H_ * 4);
  float*    hTf1  = (float*)take((size_t)B_ * H_ * 4);
  float*    hTb1  = (float*)take((size_t)B_ * H_ * 4);
  _Float16* combh = (_Float16*)take((size_t)B_ * KP_ * 2);
  _Float16* hidh  = (_Float16*)take((size_t)B_ * H_ * 2);
  const size_t HB_BYTES = (size_t)8 * 2 * 16 * H_ * 2;         // 256 KB
  char*     hbase = take(HB_BYTES + 256);
  _Float16* hbuf  = (_Float16*)hbase;
  int*      cnt   = (int*)(hbase + HB_BYTES);

  auto cvt = [&](const float* src, _Float16* dst, size_t n) {
    cvt_f16_k<<<(unsigned)((n + 255) / 256), 256, 0, stream>>>(src, dst, (int)n);
  };

  // ---- weight prep (f32 -> f16 once; fold bih+bhh; transpose code_W) ----
  for (int i = 0; i < 4; ++i) {
    cvt(WihA[i], wih_h[i], (size_t)G4_ * wihK[i]);
    cvt(WhhA[i], whh_h[i], (size_t)G4_ * H_);
    bsum_k<<<8, 256, 0, stream>>>(bihA[i], bhhA[i], bsum[i], G4_);
  }
  cvt(W2, w2h, (size_t)NC_ * H_);
  w1pad_k<<<(H_ * KP_ + 255) / 256, 256, 0, stream>>>(W1, w1p);
  trans_cw_k<<<dim3(NC_ / 16, H_ / 16), dim3(16, 16), 0, stream>>>(code_W, cWT);

  // ---- embedding (masked, coalesced gather) ----
  embed_k<<<B_ * T_, 128, 0, stream>>>(codes, lens, cWT, code_b, x0);

  // ---- layer 0: input projections (fwd + reversed gather), recurrence ----
  gemm_wmma_k<1, 8, false, false, false, true>
      <<<dim3(G4_ / 128, (B_ * T_) / 64), 128, 0, stream>>>(
          x0, H_, wih_h[0], bsum[0], xgf, B_ * T_, G4_, H_, lens);
  gemm_wmma_k<2, 8, false, false, false, true>
      <<<dim3(G4_ / 128, (B_ * T_) / 64), 128, 0, stream>>>(
          x0, H_, wih_h[1], bsum[1], xgb, B_ * T_, G4_, H_, lens);

  hipMemsetAsync(out0, 0, (size_t)B_ * T_ * 2 * H_ * 4, stream);
  hipMemsetAsync(hbase, 0, HB_BYTES + 256, stream);
  lstm_col_k<<<32, 256, 0, stream>>>(xgf, xgb, whh_h[0], whh_h[1], hbuf, cnt,
                                     out0, nullptr, nullptr, lens, 1);

  // ---- layer 1 ----
  gemm_wmma_k<1, 8, false, false, false, true>
      <<<dim3(G4_ / 128, (B_ * T_) / 64), 128, 0, stream>>>(
          out0, 2 * H_, wih_h[2], bsum[2], xgf, B_ * T_, G4_, 2 * H_, lens);
  gemm_wmma_k<2, 8, false, false, false, true>
      <<<dim3(G4_ / 128, (B_ * T_) / 64), 128, 0, stream>>>(
          out0, 2 * H_, wih_h[3], bsum[3], xgb, B_ * T_, G4_, 2 * H_, lens);

  hipMemsetAsync(hbase, 0, HB_BYTES + 256, stream);
  lstm_col_k<<<32, 256, 0, stream>>>(xgf, xgb, whh_h[2], whh_h[3], hbuf, cnt,
                                     nullptr, hTf1, hTb1, lens, 0);

  // ---- head: combined -> relu(W1) -> W2 logits ----
  comb_k<<<(B_ * KP_ + 255) / 256, 256, 0, stream>>>(hTf1, hTb1, demo, combh);
  gemm_wmma_k<0, 1, true, true, true, false>
      <<<dim3(H_ / 16, 1), 128, 0, stream>>>(
          combh, KP_, w1p, b1, hidh, B_, H_, KP_, nullptr);
  gemm_wmma_k<0, 1, true, false, false, false>
      <<<dim3(NC_ / 16, 1), 128, 0, stream>>>(
          hidh, H_, w2h, b2, d_out, B_, NC_, H_, nullptr);
}
